// BatchelorGPUNUFFTFwd_80814104642453
// MI455X (gfx1250) — matrix operand order repaired and checked by reading, not compiled, not run on hardware
//
#include <hip/hip_runtime.h>

#define NXDIM 128
#define NCOIL 8
#define NSAMP 8192
#define NTIME 4

typedef __attribute__((ext_vector_type(16))) _Float16 v16h;
typedef __attribute__((ext_vector_type(8)))  float    v8f;

// v_cos_f32 / v_sin_f32 compute trig of (2*pi * u): feed u = k*(x-64) directly.
__device__ __forceinline__ float cos2pi(float u) { return __builtin_amdgcn_cosf(u); }
__device__ __forceinline__ float sin2pi(float u) { return __builtin_amdgcn_sinf(u); }

// ---------------------------------------------------------------------------
// Kernel 1: warp image per timepoint, multiply by coil maps, convert to f16,
// and scatter into the exact per-lane WMMA B-operand layout:
//   bpack[((((tc*8 + yb)*4 + k)*2 + ri)*32 + lane)*16 + e]
// where element (lane,e) holds coil[x = 32k + 16*(lane/16) + e][y = 16*yb + lane%16].
// ---------------------------------------------------------------------------
__global__ __launch_bounds__(256) void prep_kernel(
    const float* __restrict__ image_r, const float* __restrict__ image_i,
    const float* __restrict__ csm_r,   const float* __restrict__ csm_i,
    const float* __restrict__ flow,    _Float16* __restrict__ bpack)
{
  const int tc = blockIdx.x;            // t*NCOIL + c
  const int t = tc / NCOIL, c = tc % NCOIL;
  for (int p = threadIdx.x; p < NXDIM * NXDIM; p += blockDim.x) {
    const int x = p >> 7, y = p & 127;
    const float fx = flow[((x * NXDIM + y) * 2 + 0) * NTIME + t];
    const float fy = flow[((x * NXDIM + y) * 2 + 1) * NTIME + t];
    const float cx = fminf(fmaxf((float)x + fx, 0.f), 127.f);
    const float cy = fminf(fmaxf((float)y + fy, 0.f), 127.f);
    const float x0 = floorf(cx), y0 = floorf(cy);
    const float x1 = fminf(x0 + 1.f, 127.f), y1 = fminf(y0 + 1.f, 127.f);
    const float wx = cx - x0, wy = cy - y0;
    const int x0i = (int)x0, x1i = (int)x1, y0i = (int)y0, y1i = (int)y1;
    const float w00 = (1.f - wx) * (1.f - wy), w01 = (1.f - wx) * wy;
    const float w10 = wx * (1.f - wy),         w11 = wx * wy;
    const float wr = w00 * image_r[x0i * NXDIM + y0i] + w01 * image_r[x0i * NXDIM + y1i]
                   + w10 * image_r[x1i * NXDIM + y0i] + w11 * image_r[x1i * NXDIM + y1i];
    const float wi = w00 * image_i[x0i * NXDIM + y0i] + w01 * image_i[x0i * NXDIM + y1i]
                   + w10 * image_i[x1i * NXDIM + y0i] + w11 * image_i[x1i * NXDIM + y1i];
    const float cr = csm_r[(c * NXDIM + x) * NXDIM + y];
    const float ci = csm_i[(c * NXDIM + x) * NXDIM + y];
    const float pr = wr * cr - wi * ci;
    const float pi = wr * ci + wi * cr;
    const int k = x >> 5, r2 = (x >> 4) & 1, e = x & 15;
    const int yb = y >> 4, n = y & 15, lane = r2 * 16 + n;
    const size_t base = ((((size_t)tc * 8 + yb) * 4 + k) * 2) * 512 + (size_t)lane * 16 + e;
    bpack[base]       = (_Float16)pr;    // ri = 0
    bpack[base + 512] = (_Float16)pi;    // ri = 1
  }
}

// Ey LDS swizzle: group index for (y, g) with g = row-group (rows 8g..8g+7).
// Bijective; spreads the 16 lanes of a half-wave over 16 distinct 32B groups.
__device__ __forceinline__ int ey_group(int y, int g) {
  return 2 * y + (g ^ ((y >> 3) & 1));
}

// ---------------------------------------------------------------------------
// Kernel 2: per wave = one (timepoint, 16-sample tile); 4 waves/block share t.
// Coil B-blocks (64 KB each) are DMA'd into a double-buffered LDS stage with
// global_load_async_to_lds_b128 (ASYNCcnt) while WMMAs run on the previous
// coil. A = Ex built in regs with HW 2pi trig. Complex GEMM with 3 f32 WMMA
// accumulators, fused Ey epilogue (packed f16x2, transposed + swizzled so each
// lane reads its 8 rows as one 32B chunk) + shfl_xor reduction.
// LDS: 2*64KB bbuf + 32KB Ey = 160KB -> 2 blocks / WGP.
// ---------------------------------------------------------------------------
__global__ __launch_bounds__(128) void nufft_kernel(
    const float* __restrict__ traj, const _Float16* __restrict__ bpack,
    float* __restrict__ partial)
{
  __shared__ __align__(16) _Float16 bbuf[2][32768];     // 2 x 64 KB coil stage
  __shared__ __align__(32) unsigned eyP[4][2048];       // 32 KB: [group f(y,g)][row&7], (R,I) packed

  const int lane  = threadIdx.x & 31;
  const int wv    = threadIdx.x >> 5;
  const int job   = blockIdx.x * 4 + wv;
  const int t     = job >> 9;              // 512 s-tiles per timepoint; blocks never straddle t
  const int stile = (job & 511) << 4;
  const unsigned tid = threadIdx.x;

  // --- async DMA of one coil's packed B block (64 KB) into LDS ---
  auto issue_coil = [&](int c, int buf) {
    const _Float16* src = bpack + ((size_t)(t * NCOIL + c) << 15);   // 32768 halfs per (t,c)
    const unsigned lbase = (unsigned)(uintptr_t)(&bbuf[buf][0]);     // low 32 bits = LDS offset
#pragma unroll 4
    for (int it = 0; it < 32; ++it) {                                // 128 thr * 32 * 16B = 64 KB
      const unsigned boff = (unsigned)(it * 128 + tid) * 16u;
      const unsigned la = lbase + boff;
      const unsigned long long ga = (unsigned long long)(uintptr_t)src + boff;
      asm volatile("global_load_async_to_lds_b128 %0, %1, off"
                   :: "v"(la), "v"(ga) : "memory");
    }
  };

  issue_coil(0, 0);    // start DMA of coil 0 while we do trig setup

  // --- Ey tile into LDS: entry (y, sl) at eyP[f(y, sl>>3)*8 + (sl&7)] ---
  // i = lane + 32*j  ->  sl = lane&15 (fixed per lane), y = (lane>>4) + 2*j.
  {
    const int sl = lane & 15;
    const int g = sl >> 3;
    const float ky = traj[(stile + sl) * (2 * NTIME) + NTIME + t];
    for (int i = lane; i < 16 * 128; i += 32) {
      const int y = i >> 4;
      const float u = ky * (float)(y - 64);
      union { _Float16 h[2]; unsigned u32; } pk;
      pk.h[0] = (_Float16)cos2pi(u);
      pk.h[1] = (_Float16)(-sin2pi(u));
      eyP[wv][ey_group(y, g) * 8 + (sl & 7)] = pk.u32;
    }
  }

  // --- A operand: Ex for this wave's 16 rows, K = 0..127 in 4 chunks of 32 ---
  // A layout used here (contiguous K, same order as bpack): lane m%16 = row; lane<16 holds K {0..15}+32k,
  // lane>=16 holds K {16..31}+32k; element e -> K=e+off.
  const int srow = stile + (lane & 15);
  const float kx = traj[srow * (2 * NTIME) + t];
  const int off = (lane < 16) ? 0 : 16;
  v16h axR[4], axI[4];
#pragma unroll
  for (int k = 0; k < 4; ++k) {
#pragma unroll
    for (int e = 0; e < 16; ++e) {
      const int Kloc = e + off;
      const int x = 32 * k + Kloc;
      const float u = kx * (float)(x - 64);
      axR[k][e] = (_Float16)cos2pi(u);      // Ex = cos - i sin
      axI[k][e] = (_Float16)(-sin2pi(u));
    }
  }
  __syncthreads();   // Ey tiles visible (bbuf[0] not read before the async wait below)

  const int halfsel = lane >> 4;   // C layout: rows v (lanes<16) / v+8 (lanes>=16)
  const int col = lane & 15;       // C layout: column = lane%16

  for (int c = 0; c < NCOIL; ++c) {
    const int cur = c & 1;
    if (c < NCOIL - 1) {
      issue_coil(c + 1, cur ^ 1);                          // prefetch next coil
      asm volatile("s_wait_asynccnt 0x20" ::: "memory");   // batch c (older 32) complete
    } else {
      asm volatile("s_wait_asynccnt 0x0" ::: "memory");
    }
    __syncthreads();   // all waves' DMA parts of coil c landed in bbuf[cur]

    float ksR[8], ksI[8];
#pragma unroll
    for (int v = 0; v < 8; ++v) { ksR[v] = 0.f; ksI[v] = 0.f; }

    for (int yb = 0; yb < 8; ++yb) {
      v8f accRp = {}, accRn = {}, accI = {};
      const int cb = yb * 4096;                            // (yb*4 + k=0, ri=0) * 512 halfs
#pragma unroll
      for (int k = 0; k < 4; ++k) {
        const v16h bR = *(const v16h*)&bbuf[cur][cb + k * 1024 + lane * 16];
        const v16h bI = *(const v16h*)&bbuf[cur][cb + k * 1024 + 512 + lane * 16];
        accRp = __builtin_amdgcn_wmma_f32_16x16x32_f16(false, axR[k], false, bR, (short)0, accRp, false, false);
        accRn = __builtin_amdgcn_wmma_f32_16x16x32_f16(false, axI[k], false, bI, (short)0, accRn, false, false);
        accI  = __builtin_amdgcn_wmma_f32_16x16x32_f16(false, axR[k], false, bI, (short)0, accI,  false, false);
        accI  = __builtin_amdgcn_wmma_f32_16x16x32_f16(false, axI[k], false, bR, (short)0, accI,  false, false);
      }
      // fused Ey epilogue: one 32B LDS chunk = this lane's 8 rows (R,I packed)
      const int y = yb * 16 + col;
      const v16h eyv = *(const v16h*)&eyP[wv][ey_group(y, halfsel) * 8];
#pragma unroll
      for (int v = 0; v < 8; ++v) {
        const float tR = accRp[v] - accRn[v];
        const float tI = accI[v];
        const float eR = (float)eyv[2 * v];
        const float eI = (float)eyv[2 * v + 1];
        ksR[v] += tR * eR - tI * eI;
        ksI[v] += tR * eI + tI * eR;
      }
    }
    __syncthreads();   // done reading bbuf[cur]; it may be DMA-retargeted at stage c+2

    // reduce across the 16 columns held by each half-wave
#pragma unroll
    for (int v = 0; v < 8; ++v) {
#pragma unroll
      for (int m = 1; m < 16; m <<= 1) {
        ksR[v] += __shfl_xor(ksR[v], m);
        ksI[v] += __shfl_xor(ksI[v], m);
      }
    }
    if ((lane & 15) == 0) {
#pragma unroll
      for (int v = 0; v < 8; ++v) {
        const int s = stile + v + 8 * halfsel;
        const size_t o = ((size_t)(t * NCOIL + c) * NSAMP + s) * 2;
        partial[o]     = ksR[v];
        partial[o + 1] = ksI[v];
      }
    }
  }
}

// ---------------------------------------------------------------------------
// Kernel 3: sum over timepoints -> d_out (complex interleaved [Nc, NS]).
// ---------------------------------------------------------------------------
__global__ __launch_bounds__(256) void reduce_kernel(
    const float* __restrict__ partial, float* __restrict__ out)
{
  const int i = blockIdx.x * blockDim.x + threadIdx.x;   // c*NSAMP + s
  if (i >= NCOIL * NSAMP) return;
  float r = 0.f, im = 0.f;
#pragma unroll
  for (int t = 0; t < NTIME; ++t) {
    const size_t o = ((size_t)t * NCOIL * NSAMP + i) * 2;
    r  += partial[o];
    im += partial[o + 1];
  }
  out[(size_t)i * 2]     = r;
  out[(size_t)i * 2 + 1] = im;
}

extern "C" void kernel_launch(void* const* d_in, const int* in_sizes, int n_in,
                              void* d_out, int out_size, void* d_ws, size_t ws_size,
                              hipStream_t stream)
{
  const float* image_r = (const float*)d_in[0];
  const float* image_i = (const float*)d_in[1];
  const float* csm_r   = (const float*)d_in[2];
  const float* csm_i   = (const float*)d_in[3];
  const float* traj    = (const float*)d_in[4];
  // d_in[5] = dcf: unused by the reference forward model
  const float* flow    = (const float*)d_in[6];

  _Float16* bpack = (_Float16*)d_ws;
  const size_t bpack_bytes = (size_t)NTIME * NCOIL * NXDIM * NXDIM * 2 * sizeof(_Float16); // 2 MB
  float* partial = (float*)((char*)d_ws + bpack_bytes);                                    // 2 MB

  hipLaunchKernelGGL(prep_kernel, dim3(NTIME * NCOIL), dim3(256), 0, stream,
                     image_r, image_i, csm_r, csm_i, flow, bpack);
  hipLaunchKernelGGL(nufft_kernel, dim3(NTIME * (NSAMP / 16) / 4), dim3(128), 0, stream,
                     traj, bpack, partial);
  hipLaunchKernelGGL(reduce_kernel, dim3((NCOIL * NSAMP + 255) / 256), dim3(256), 0, stream,
                     partial, (float*)d_out);
}
